// TransformerNet_19404662243712
// MI455X (gfx1250) — matrix operand order repaired and checked by reading, not compiled, not run on hardware
//
#include <hip/hip_runtime.h>
#include <math.h>

// ---------------------------------------------------------------------------
// TransformerConv x2 + Linear for gfx1250 (MI455X), wave32 + WMMA f16.
// ---------------------------------------------------------------------------

typedef __attribute__((ext_vector_type(16))) _Float16 v16h;
typedef __attribute__((ext_vector_type(8)))  float    v8f;

#define HEADS 4
#define DHEAD 64
#define HD    256   // HEADS*DHEAD

// ---------------------------------------------------------------------------
// Zero-fill (graph-capture safe scratch init)
// ---------------------------------------------------------------------------
__global__ void tnet_zero_f32(float* __restrict__ p, long long n) {
  long long i = (long long)blockIdx.x * blockDim.x + threadIdx.x;
  if (i < n) p[i] = 0.0f;
}

// ---------------------------------------------------------------------------
// GEMM with bias: out[n,M] = A[n,K] @ W[K,M] + b[M]
// One wave32 per 16x16 tile, K stepped by 32, v_wmma_f32_16x16x32_f16.
// A fragment (contiguous k order, same as B): lane m = l&15, half = l>>4,
//   a[0..7]  = A[m][kBase + 16*half + 0..7]
//   a[8..15] = A[m][kBase + 16*half + 8 + 0..7]
// B fragment (32x16, row striped across lanes): lane col = l&15, half = l>>4,
//   b[2r+j]  = W[kBase + 16*half + 2r + j][col]
// C/D: VGPR g, lane l -> row = g + 8*(l>>4), col = l&15.
// COLGUARD=false fast path: zero per-lane branches; store epilogue uses a
// wave-uniform full-tile path with strength-reduced int offsets.
// All early exits are wave-uniform => EXEC all-ones at WMMA (ISA requirement).
// All index math is 32-bit (n*M, n*K < 2^23 for this net).
// ---------------------------------------------------------------------------
template <bool COLGUARD>
__global__ void tnet_gemm_bias_wmma(const float* __restrict__ A,
                                    const float* __restrict__ W,
                                    const float* __restrict__ bias,
                                    float* __restrict__ out,
                                    int n, int K, int M) {
  const int lane     = threadIdx.x;              // 0..31
  const int rowTile  = blockIdx.x;
  const int colTile  = blockIdx.y * 4 + threadIdx.y;
  const int nColTile = (M + 15) >> 4;
  if (colTile >= nColTile) return;               // wave-uniform
  const int row0 = rowTile << 4;
  if (row0 >= n) return;                         // wave-uniform

  const int m    = lane & 15;
  const int half = lane >> 4;
  const int row  = row0 + m;                     // A row for this lane
  const int col  = (colTile << 4) + m;           // B/C column for this lane
  const bool colOK = (!COLGUARD) || (col < M);

  // Per-lane base pointers, hoisted out of the K loop (int offsets).
  const float* aRow = A + row * K;
  const float* wCol = W + (colOK ? col : 0);     // safe dummy column

  v8f c = {};
  for (int kBase = 0; kBase < K; kBase += 32) {
    // ---- A fragment: two contiguous 8-float runs -> float4 x4 loads ----
    v16h a;
    const float* ap0 = aRow + kBase + 16 * half;
    const float* ap1 = ap0 + 8;
#pragma unroll
    for (int j = 0; j < 8; ++j) {
      a[j]     = (_Float16)ap0[j];
      a[8 + j] = (_Float16)ap1[j];
    }
    // ---- B fragment: 16 strided loads along K (32-bit index math) ----
    v16h b;
    const int kOff = (kBase + 16 * half) * M;
#pragma unroll
    for (int r = 0; r < 8; ++r) {
      float b0 = wCol[kOff + 2 * r * M];
      float b1 = wCol[kOff + (2 * r + 1) * M];
      if (COLGUARD && !colOK) { b0 = 0.0f; b1 = 0.0f; }
      b[2 * r]     = (_Float16)b0;
      b[2 * r + 1] = (_Float16)b1;
    }
    c = __builtin_amdgcn_wmma_f32_16x16x32_f16(
        /*neg_a=*/false, a, /*neg_b=*/false, b,
        /*c_mod=*/(short)0, c, /*reuse_a=*/false, /*reuse_b=*/false);
  }

  // ---- Store epilogue ----
  const float bv = colOK ? bias[col] : 0.0f;
  // Lane's first output element: row (row0 + 8*half), column col; rows step by 1
  // => offset steps by M. int offsets only (no 64-bit muls).
  float* oBase = out + (row0 + 8 * half) * M + (colOK ? col : 0);

  if (row0 + 16 <= n) {                          // wave-uniform: full tile
    if (colOK) {                                 // single exec toggle (guarded
#pragma unroll                                   // variant only; no-op in fast)
      for (int g = 0; g < 8; ++g) oBase[g * M] = c[g] + bv;
    }
  } else {                                       // partial row tile (not hit
#pragma unroll                                   // for N=20000, kept for safety)
    for (int g = 0; g < 8; ++g) {
      const int orow = row0 + 8 * half + g;
      if (colOK && orow < n) oBase[g * M] = c[g] + bv;
    }
  }
}

// ---------------------------------------------------------------------------
// Per-edge attention logits: alpha[e,h] = SCALE * dot(q[dst,h,:], k[src,h,:])
// One wave per edge; lane covers 8 of the 256 channels; 8-lane tree reduce.
// ---------------------------------------------------------------------------
__global__ void tnet_edge_logits(const float* __restrict__ q,
                                 const float* __restrict__ k,
                                 const int* __restrict__ ei,
                                 float* __restrict__ alpha,
                                 int E, float scale) {
  const int wave = blockIdx.x * (blockDim.x >> 5) + (threadIdx.x >> 5);
  const int lane = threadIdx.x & 31;
  if (wave >= E) return;
  const int s = ei[wave];
  const int d = ei[E + wave];
  const float4* qp = (const float4*)(q + (long long)d * HD + lane * 8);
  const float4* kp = (const float4*)(k + (long long)s * HD + lane * 8);
  const float4 q0 = qp[0], q1 = qp[1], k0 = kp[0], k1 = kp[1];
  float p = q0.x * k0.x + q0.y * k0.y + q0.z * k0.z + q0.w * k0.w
          + q1.x * k1.x + q1.y * k1.y + q1.z * k1.z + q1.w * k1.w;
  p += __shfl_xor(p, 1, 32);
  p += __shfl_xor(p, 2, 32);
  p += __shfl_xor(p, 4, 32);
  if ((lane & 7) == 0)
    alpha[(long long)wave * HEADS + (lane >> 3)] = p * scale;
}

// Order-preserving float<->uint encoding for atomic segment-max.
__device__ __forceinline__ unsigned tnet_f2ord(float f) {
  unsigned u = __float_as_uint(f);
  return (u & 0x80000000u) ? ~u : (u | 0x80000000u);
}
__device__ __forceinline__ float tnet_ord2f(unsigned e) {
  unsigned u = (e & 0x80000000u) ? (e & 0x7FFFFFFFu) : ~e;
  return __uint_as_float(u);
}

__global__ void tnet_seg_max(const float* __restrict__ alpha,
                             const int* __restrict__ ei,
                             unsigned* __restrict__ amax, int E) {
  long long i = (long long)blockIdx.x * blockDim.x + threadIdx.x;
  if (i >= (long long)E * HEADS) return;
  const int e = (int)(i >> 2);
  const int h = (int)(i & 3);
  const int d = ei[E + e];
  atomicMax(&amax[d * HEADS + h], tnet_f2ord(alpha[i]));
}

__global__ void tnet_seg_exp_sum(float* __restrict__ alpha,
                                 const int* __restrict__ ei,
                                 const unsigned* __restrict__ amax,
                                 float* __restrict__ den, int E) {
  long long i = (long long)blockIdx.x * blockDim.x + threadIdx.x;
  if (i >= (long long)E * HEADS) return;
  const int e = (int)(i >> 2);
  const int h = (int)(i & 3);
  const int d = ei[E + e];
  const float m  = tnet_ord2f(amax[d * HEADS + h]);  // finite: d has >=1 edge
  const float ex = expf(alpha[i] - m);
  alpha[i] = ex;                                     // reuse buffer for exp()
  atomicAdd(&den[d * HEADS + h], ex);
}

// ---------------------------------------------------------------------------
// Weighted scatter: agg[dst,h,:] += (ex/den[dst,h]) * v[src,h,:]
// One wave per edge; f32 atomics land in L2 (agg is ~20MB, L2-resident).
// ---------------------------------------------------------------------------
__global__ void tnet_edge_aggregate(const float* __restrict__ v,
                                    const float* __restrict__ exw,
                                    const float* __restrict__ den,
                                    const int* __restrict__ ei,
                                    float* __restrict__ agg, int E) {
  const int wave = blockIdx.x * (blockDim.x >> 5) + (threadIdx.x >> 5);
  const int lane = threadIdx.x & 31;
  if (wave >= E) return;
  const int s = ei[wave];
  const int d = ei[E + wave];
  const int h = lane >> 3;
  const float w = exw[(long long)wave * HEADS + h] / den[d * HEADS + h];
  const float* vp = v   + (long long)s * HD + lane * 8;
  float*       ap = agg + (long long)d * HD + lane * 8;
#pragma unroll
  for (int j = 0; j < 8; ++j) atomicAdd(&ap[j], w * vp[j]);
}

// ---------------------------------------------------------------------------
// out[n,d] = relu( mean_h agg[n,h,d] + skip[n,d] )
// ---------------------------------------------------------------------------
__global__ void tnet_finalize(const float* __restrict__ agg,
                              const float* __restrict__ skip,
                              float* __restrict__ out, int n, int doRelu) {
  long long i = (long long)blockIdx.x * blockDim.x + threadIdx.x;
  if (i >= (long long)n * DHEAD) return;
  const int node = (int)(i >> 6);
  const int dd   = (int)(i & 63);
  const float* ap = agg + (long long)node * HD + dd;
  float s = (ap[0] + ap[64] + ap[128] + ap[192]) * 0.25f + skip[i];
  out[i] = doRelu ? fmaxf(s, 0.0f) : s;
}

// ---------------------------------------------------------------------------
// Host-side orchestration
// ---------------------------------------------------------------------------
static void tnet_run_gemm(const float* A, const float* W, const float* b,
                          float* out, int n, int K, int M, hipStream_t stream) {
  dim3 blk(32, 4);
  dim3 grd((n + 15) / 16, (((M + 15) / 16) + 3) / 4);
  if ((M & 15) == 0)
    tnet_gemm_bias_wmma<false><<<grd, blk, 0, stream>>>(A, W, b, out, n, K, M);
  else
    tnet_gemm_bias_wmma<true><<<grd, blk, 0, stream>>>(A, W, b, out, n, K, M);
}

extern "C" void kernel_launch(void* const* d_in, const int* in_sizes, int n_in,
                              void* d_out, int out_size, void* d_ws, size_t ws_size,
                              hipStream_t stream) {
  const float* x   = (const float*)d_in[0];
  const int*   ei  = (const int*)d_in[1];   // [2,E]: row0 = src, row1 = dst
  const float* Wq1 = (const float*)d_in[2],  *bq1 = (const float*)d_in[3];
  const float* Wk1 = (const float*)d_in[4],  *bk1 = (const float*)d_in[5];
  const float* Wv1 = (const float*)d_in[6],  *bv1 = (const float*)d_in[7];
  const float* Ws1 = (const float*)d_in[8],  *bs1 = (const float*)d_in[9];
  const float* Wq2 = (const float*)d_in[10], *bq2 = (const float*)d_in[11];
  const float* Wk2 = (const float*)d_in[12], *bk2 = (const float*)d_in[13];
  const float* Wv2 = (const float*)d_in[14], *bv2 = (const float*)d_in[15];
  const float* Ws2 = (const float*)d_in[16], *bs2 = (const float*)d_in[17];
  const float* Wl  = (const float*)d_in[18], *bl  = (const float*)d_in[19];
  float* out = (float*)d_out;

  const int IN = 128;
  const int N  = in_sizes[0] / IN;
  const int E  = in_sizes[1] / 2;
  const int C  = in_sizes[19];
  const float scale = 0.125f;  // 1/sqrt(64)

  // Workspace carve-up (floats). agg/amax/den contiguous for one zero pass.
  float* ws = (float*)d_ws;
  size_t off = 0;
  float* q    = ws + off; off += (size_t)N * HD;
  float* k    = ws + off; off += (size_t)N * HD;
  float* v    = ws + off; off += (size_t)N * HD;
  float* skip = ws + off; off += (size_t)N * DHEAD;
  float* agg  = ws + off; off += (size_t)N * HD;       // |
  unsigned* amax = (unsigned*)(ws + off); off += (size_t)N * HEADS; // | zeroed
  float* den  = ws + off; off += (size_t)N * HEADS;    // | together
  float* alpha= ws + off; off += (size_t)E * HEADS;
  float* h1   = ws + off; off += (size_t)N * DHEAD;
  float* h2   = ws + off; off += (size_t)N * DHEAD;
  (void)ws_size; (void)n_in; (void)out_size;

  const int edgeBlocks = (E + 7) / 8;           // 8 waves (edges) per block
  const long long eh = (long long)E * HEADS;
  const long long zcount = (long long)N * (HD + 2 * HEADS); // agg+amax+den
  const long long fcount = (long long)N * DHEAD;

  // ---- Layer 1 (K = IN = 128) ----
  tnet_run_gemm(x, Wq1, bq1, q,    N, IN, HD,    stream);
  tnet_run_gemm(x, Wk1, bk1, k,    N, IN, HD,    stream);
  tnet_run_gemm(x, Wv1, bv1, v,    N, IN, HD,    stream);
  tnet_run_gemm(x, Ws1, bs1, skip, N, IN, DHEAD, stream);
  tnet_zero_f32<<<(int)((zcount + 255) / 256), 256, 0, stream>>>(agg, zcount);
  tnet_edge_logits<<<edgeBlocks, 256, 0, stream>>>(q, k, ei, alpha, E, scale);
  tnet_seg_max<<<(int)((eh + 255) / 256), 256, 0, stream>>>(alpha, ei, amax, E);
  tnet_seg_exp_sum<<<(int)((eh + 255) / 256), 256, 0, stream>>>(alpha, ei, amax, den, E);
  tnet_edge_aggregate<<<edgeBlocks, 256, 0, stream>>>(v, alpha, den, ei, agg, E);
  tnet_finalize<<<(int)((fcount + 255) / 256), 256, 0, stream>>>(agg, skip, h1, N, 1);

  // ---- Layer 2 (K = DHEAD = 64) ----
  tnet_run_gemm(h1, Wq2, bq2, q,    N, DHEAD, HD,    stream);
  tnet_run_gemm(h1, Wk2, bk2, k,    N, DHEAD, HD,    stream);
  tnet_run_gemm(h1, Wv2, bv2, v,    N, DHEAD, HD,    stream);
  tnet_run_gemm(h1, Ws2, bs2, skip, N, DHEAD, DHEAD, stream);
  tnet_zero_f32<<<(int)((zcount + 255) / 256), 256, 0, stream>>>(agg, zcount);
  tnet_edge_logits<<<edgeBlocks, 256, 0, stream>>>(q, k, ei, alpha, E, scale);
  tnet_seg_max<<<(int)((eh + 255) / 256), 256, 0, stream>>>(alpha, ei, amax, E);
  tnet_seg_exp_sum<<<(int)((eh + 255) / 256), 256, 0, stream>>>(alpha, ei, amax, den, E);
  tnet_edge_aggregate<<<edgeBlocks, 256, 0, stream>>>(v, alpha, den, ei, agg, E);
  tnet_finalize<<<(int)((fcount + 255) / 256), 256, 0, stream>>>(agg, skip, h2, N, 1);

  // ---- Classifier: out[N,C] = h2 @ Wl + bl  (C=40, column-guarded) ----
  tnet_run_gemm(h2, Wl, bl, out, N, DHEAD, C, stream);
}
